// socialRecModel_67602785239413
// MI455X (gfx1250) — matrix operand order repaired and checked by reading, not compiled, not run on hardware
//
#include <hip/hip_runtime.h>

// ---------------------------------------------------------------------------
// MI455X (gfx1250) fully-fused diffusion-MLP, wave32 + v_wmma_f32_16x16x32_bf16:
//   t_emb = (cos|sin)(t*freqs) @ W_step^T + b_step     (WMMA, LDS-resident)
//   out   = leakyrelu([x|c|t_emb] @ W1^T + b1) @ W2^T + b2
// One workgroup = 64 batch rows; emb/input/hidden tiles all live in LDS
// (~194 KB of 320 KB WGP LDS).  K-outermost loop nest: A-fragments read from
// LDS once per k-step; B-fragments (weights, L2-resident) are software-
// pipelined one full k-step ahead to cover L2 latency.
// ---------------------------------------------------------------------------

typedef __attribute__((ext_vector_type(16))) __bf16 v16bf;
typedef __attribute__((ext_vector_type(8)))  __bf16 v8bf;
typedef __attribute__((ext_vector_type(4)))  __bf16 v4bf;
typedef __attribute__((ext_vector_type(8)))  float  v8f;
typedef __attribute__((ext_vector_type(4)))  float  v4f;

#define D        256
#define D3       768
#define ROWS     64                 // batch rows per workgroup
#define PAD      8                  // LDS row padding (bf16 elems)
#define LDSTR    (D3 + PAD)         // 776 elems, 1552 B, 16B-aligned rows
#define ESTR     (D  + PAD)         // 264 elems for embedding tile
#define NTHREADS 256                // 8 waves

static __device__ inline v16bf combine16(v8bf lo, v8bf hi) {
    return __builtin_shufflevector(lo, hi, 0, 1, 2, 3, 4, 5, 6, 7,
                                           8, 9, 10, 11, 12, 13, 14, 15);
}

// A fragment (16x32 bf16) from an LDS tile, row-major, stride ldk (elems).
static __device__ inline v16bf load_A_lds(const __bf16* base, int ldk,
                                          int m0, int k0, int lane) {
    int m     = m0 + (lane & 15);
    int half8 = (lane >> 4) << 3;
    const __bf16* p = base + m * ldk + k0 + half8;
    v8bf lo = *(const v8bf*)(p);
    v8bf hi = *(const v8bf*)(p + 16);
    return combine16(lo, hi);
}

// B fragment (32x16 bf16) for out = act @ W^T:  B[k][n] = W[n0+n][k0+k].
static __device__ inline v16bf load_B_w(const __bf16* __restrict__ W, int ldk,
                                        int n0, int k0, int lane) {
    int n  = n0 + (lane & 15);
    int kb = k0 + ((lane >> 4) << 3);
    const __bf16* p = W + n * ldk + kb;
    v8bf lo = *(const v8bf*)(p);
    v8bf hi = *(const v8bf*)(p + 16);
    return combine16(lo, hi);
}

static __device__ inline v8f wmma_bf16(v16bf a, v16bf b, v8f c) {
    return __builtin_amdgcn_wmma_f32_16x16x32_bf16(
        false, a, false, b, (short)0, c, false, false);
}

// ---------------------------------------------------------------------------
// Kernel 0: fp32 -> bf16 weight conversion (tiny, one-shot)
// ---------------------------------------------------------------------------
__global__ void cvt_bf16_kernel(const float* __restrict__ src,
                                __bf16* __restrict__ dst, int n) {
    int i = blockIdx.x * blockDim.x + threadIdx.x;
    if (i < n) dst[i] = (__bf16)src[i];
}

// ---------------------------------------------------------------------------
// Fused kernel: embedding GEMM + layer1 + LeakyReLU + layer2
// ---------------------------------------------------------------------------
__global__ __launch_bounds__(NTHREADS) void fused_mlp_kernel(
    const float*  __restrict__ x,        // [B,256] f32
    const float*  __restrict__ c,        // [B,256] f32
    const int*    __restrict__ t,        // [B]
    const __bf16* __restrict__ Wstepb,   // [256,256] bf16
    const float*  __restrict__ b_step,   // [256]
    const __bf16* __restrict__ W1b,      // [768,768] bf16
    const float*  __restrict__ b1,       // [768]
    const __bf16* __restrict__ W2b,      // [256,768] bf16
    const float*  __restrict__ b2,       // [256]
    float*        __restrict__ out)      // [B,256] f32
{
    extern __shared__ __bf16 smem[];
    __bf16* mi = smem;                   // 64 x 776  input tile [x|c|t_emb]
    __bf16* hs = smem + ROWS * LDSTR;    // 64 x 776  hidden tile
    __bf16* es = hs;                     // 64 x 264  embedding tile (aliases hs)

    const int row0 = blockIdx.x * ROWS;
    const int tid  = threadIdx.x;
    const int wave = tid >> 5, lane = tid & 31;
    const int mofs = (lane >> 4) << 3;   // C/D row offset for upper lane half

    // ---- Stage A: stage x|c into mi (bf16) and sinusoid tile into es ----
    for (int idx = tid; idx < ROWS * (2 * D) / 4; idx += NTHREADS) {
        int e   = idx * 4;
        int r   = e >> 9;                 // / 512
        int col = e & 511;
        size_t g = (size_t)(row0 + r) * D;
        v4f f = (col < 256) ? *(const v4f*)(x + g + col)
                            : *(const v4f*)(c + g + (col - 256));
        v4bf v;
#pragma unroll
        for (int i = 0; i < 4; ++i) v[i] = (__bf16)f[i];
        *(v4bf*)(mi + r * LDSTR + col) = v;
    }
    const float LOGMP_OVER_HALF = 0.07195051793294886f;  // ln(10000)/128
    for (int idx = tid; idx < ROWS * D; idx += NTHREADS) {
        int r   = idx >> 8;
        int col = idx & 255;
        float tv   = (float)t[row0 + r];
        int   j    = col & 127;
        float freq = __expf(-LOGMP_OVER_HALF * (float)j);
        float a    = tv * freq;
        float v    = (col < 128) ? cosf(a) : sinf(a);
        es[r * ESTR + col] = (__bf16)v;
    }
    __syncthreads();

    // ---- Stage A2: t_emb = es @ Wstep^T + b_step -> mi[:, 512:768] ----
    {
        v8f acc[2][4];
#pragma unroll
        for (int nb = 0; nb < 2; ++nb)
#pragma unroll
            for (int mr = 0; mr < 4; ++mr) acc[nb][mr] = v8f{};
        v16bf bcur[2], bnext[2];
#pragma unroll
        for (int nb = 0; nb < 2; ++nb)
            bcur[nb] = load_B_w(Wstepb, D, wave * 32 + nb * 16, 0, lane);
        for (int k0 = 0; k0 < D; k0 += 32) {
            const int kn = (k0 + 32 < D) ? k0 + 32 : k0;
#pragma unroll
            for (int nb = 0; nb < 2; ++nb)
                bnext[nb] = load_B_w(Wstepb, D, wave * 32 + nb * 16, kn, lane);
            v16bf a[4];
#pragma unroll
            for (int mr = 0; mr < 4; ++mr)
                a[mr] = load_A_lds(es, ESTR, mr * 16, k0, lane);
#pragma unroll
            for (int nb = 0; nb < 2; ++nb)
#pragma unroll
                for (int mr = 0; mr < 4; ++mr)
                    acc[nb][mr] = wmma_bf16(a[mr], bcur[nb], acc[nb][mr]);
#pragma unroll
            for (int nb = 0; nb < 2; ++nb) bcur[nb] = bnext[nb];
        }
        __syncthreads();   // done reading es before overwriting hs region
#pragma unroll
        for (int nb = 0; nb < 2; ++nb) {
            const int n0     = wave * 32 + nb * 16;
            const float bias = b_step[n0 + (lane & 15)];
#pragma unroll
            for (int mr = 0; mr < 4; ++mr)
#pragma unroll
                for (int r = 0; r < 8; ++r) {
                    int m = mr * 16 + r + mofs;
                    mi[m * LDSTR + 512 + n0 + (lane & 15)] =
                        (__bf16)(acc[nb][mr][r] + bias);
                }
        }
    }
    __syncthreads();

    // ---- Stage B: h = leakyrelu(mi @ W1^T + b1) -> hs ----
    {
        v8f acc[6][4];
#pragma unroll
        for (int nb = 0; nb < 6; ++nb)
#pragma unroll
            for (int mr = 0; mr < 4; ++mr) acc[nb][mr] = v8f{};
        v16bf bcur[6], bnext[6];
#pragma unroll
        for (int nb = 0; nb < 6; ++nb)
            bcur[nb] = load_B_w(W1b, D3, wave * 96 + nb * 16, 0, lane);
        for (int k0 = 0; k0 < D3; k0 += 32) {
            const int kn = (k0 + 32 < D3) ? k0 + 32 : k0;
#pragma unroll
            for (int nb = 0; nb < 6; ++nb)
                bnext[nb] = load_B_w(W1b, D3, wave * 96 + nb * 16, kn, lane);
            v16bf a[4];
#pragma unroll
            for (int mr = 0; mr < 4; ++mr)
                a[mr] = load_A_lds(mi, LDSTR, mr * 16, k0, lane);
#pragma unroll
            for (int nb = 0; nb < 6; ++nb)
#pragma unroll
                for (int mr = 0; mr < 4; ++mr)
                    acc[nb][mr] = wmma_bf16(a[mr], bcur[nb], acc[nb][mr]);
#pragma unroll
            for (int nb = 0; nb < 6; ++nb) bcur[nb] = bnext[nb];
        }
#pragma unroll
        for (int nb = 0; nb < 6; ++nb) {
            const int n0     = wave * 96 + nb * 16;
            const float bias = b1[n0 + (lane & 15)];
#pragma unroll
            for (int mr = 0; mr < 4; ++mr)
#pragma unroll
                for (int r = 0; r < 8; ++r) {
                    float v = acc[nb][mr][r] + bias;
                    v = (v > 0.0f) ? v : 0.01f * v;      // LeakyReLU(0.01)
                    int m = mr * 16 + r + mofs;
                    hs[m * LDSTR + n0 + (lane & 15)] = (__bf16)v;
                }
        }
    }
    __syncthreads();

    // ---- Stage C: out = hs @ W2^T + b2 ----
    {
        v8f acc[2][4];
#pragma unroll
        for (int nb = 0; nb < 2; ++nb)
#pragma unroll
            for (int mr = 0; mr < 4; ++mr) acc[nb][mr] = v8f{};
        v16bf bcur[2], bnext[2];
#pragma unroll
        for (int nb = 0; nb < 2; ++nb)
            bcur[nb] = load_B_w(W2b, D3, wave * 32 + nb * 16, 0, lane);
        for (int k0 = 0; k0 < D3; k0 += 32) {
            const int kn = (k0 + 32 < D3) ? k0 + 32 : k0;
#pragma unroll
            for (int nb = 0; nb < 2; ++nb)
                bnext[nb] = load_B_w(W2b, D3, wave * 32 + nb * 16, kn, lane);
            v16bf a[4];
#pragma unroll
            for (int mr = 0; mr < 4; ++mr)
                a[mr] = load_A_lds(hs, LDSTR, mr * 16, k0, lane);
#pragma unroll
            for (int nb = 0; nb < 2; ++nb)
#pragma unroll
                for (int mr = 0; mr < 4; ++mr)
                    acc[nb][mr] = wmma_bf16(a[mr], bcur[nb], acc[nb][mr]);
#pragma unroll
            for (int nb = 0; nb < 2; ++nb) bcur[nb] = bnext[nb];
        }
#pragma unroll
        for (int nb = 0; nb < 2; ++nb) {
            const int n0     = wave * 32 + nb * 16;
            const float bias = b2[n0 + (lane & 15)];
#pragma unroll
            for (int mr = 0; mr < 4; ++mr)
#pragma unroll
                for (int r = 0; r < 8; ++r) {
                    int m = mr * 16 + r + mofs;
                    out[(size_t)(row0 + m) * D + n0 + (lane & 15)] =
                        acc[nb][mr][r] + bias;
                }
        }
    }
}

// ---------------------------------------------------------------------------
// Host launcher.  inputs: 0:x 1:c 2:t 3:W_step 4:b_step 5:W1 6:b1 7:W2 8:b2
// ---------------------------------------------------------------------------
extern "C" void kernel_launch(void* const* d_in, const int* in_sizes, int n_in,
                              void* d_out, int out_size, void* d_ws, size_t ws_size,
                              hipStream_t stream) {
    const float* x      = (const float*)d_in[0];
    const float* c      = (const float*)d_in[1];
    const int*   t      = (const int*)  d_in[2];
    const float* W_step = (const float*)d_in[3];
    const float* b_step = (const float*)d_in[4];
    const float* W1     = (const float*)d_in[5];
    const float* b1     = (const float*)d_in[6];
    const float* W2     = (const float*)d_in[7];
    const float* b2     = (const float*)d_in[8];
    float* out = (float*)d_out;

    const int B = in_sizes[0] / D;   // 131072

    // Workspace: bf16 weights only (~1.7 MB)
    char* ws = (char*)d_ws;
    __bf16* Wstep_b = (__bf16*)ws;
    size_t  off     = (size_t)D * D * sizeof(__bf16);
    __bf16* W1_b    = (__bf16*)(ws + off);  off += (size_t)D3 * D3 * sizeof(__bf16);
    __bf16* W2_b    = (__bf16*)(ws + off);

    {
        int n0 = D * D, n1 = D3 * D3, n2 = D * D3;
        cvt_bf16_kernel<<<(n0 + 255) / 256, 256, 0, stream>>>(W_step, Wstep_b, n0);
        cvt_bf16_kernel<<<(n1 + 255) / 256, 256, 0, stream>>>(W1,     W1_b,    n1);
        cvt_bf16_kernel<<<(n2 + 255) / 256, 256, 0, stream>>>(W2,     W2_b,    n2);
    }

    const size_t smem_bytes = (size_t)2 * ROWS * LDSTR * sizeof(__bf16); // ~194 KB
    fused_mlp_kernel<<<B / ROWS, NTHREADS, smem_bytes, stream>>>(
        x, c, t, Wstep_b, b_step, W1_b, b1, W2_b, b2, out);
}
